// VQ_24670292148591
// MI455X (gfx1250) — matrix operand order repaired and checked by reading, not compiled, not run on hardware
//
#include <hip/hip_runtime.h>
#include <hip/hip_bf16.h>

typedef __attribute__((ext_vector_type(16))) __bf16 v16bf;
typedef __attribute__((ext_vector_type(8)))  __bf16 v8bf;
typedef __attribute__((ext_vector_type(8)))  float  v8f;
typedef __attribute__((ext_vector_type(4)))  float  v4f;

#define B_   4
#define D_   512
#define T_   4096
#define K_   4096
#define BM   64          // rows (t) per workgroup
#define BN   128         // codes per K-chunk
#define DS   128         // d-slice of the codebook chunk staged in LDS
#define LDA  520         // 512 + 8 bf16 pad (16B-aligned rows, spreads banks)
#define LDB  136         // 128 + 8 bf16 pad
#define NWAVES 8
#define BLOCK  256
#define NBLOCKS (B_ * T_ / BM)   // 256 workgroups
#define NKC    (K_ / BN)         // 32 codebook chunks
#define BUFELE (2 * BN * LDB)    // one ping-pong buffer (hi+lo), in bf16 elements

// ---------------- Kernel 1: half squared norms of codebook rows ----------------
__global__ void vq_norms(const float* __restrict__ emb, float* __restrict__ halfn) {
    int wave = threadIdx.x >> 5;
    int lane = threadIdx.x & 31;
    int row  = blockIdx.x * (BLOCK / 32) + wave;
    const float* e = emb + (size_t)row * D_;
    float s = 0.f;
    #pragma unroll 4
    for (int i = lane; i < D_; i += 32) { float v = e[i]; s += v * v; }
    #pragma unroll
    for (int off = 16; off > 0; off >>= 1) s += __shfl_xor(s, off, 32);
    if (lane == 0) halfn[row] = 0.5f * s;
}

#define WMMA_BF16(A, Bv, C) \
    __builtin_amdgcn_wmma_f32_16x16x32_bf16(false, (A), false, (Bv), (short)0, (C), false, false)

__device__ __forceinline__ v16bf ldA(const __bf16* p) {   // A frag: 2 x 8 bf16, 16B apart (16 contiguous k)
    union { v16bf v; v8bf h[2]; } u;
    u.h[0] = *(const v8bf*)(p);
    u.h[1] = *(const v8bf*)(p + 8);
    return u.v;
}

// stage a BN x DS slice of the codebook into LDS as bf16 hi/lo (coalesced)
__device__ __forceinline__ void stage_slice(const float* __restrict__ emb, int kbase,
                                            int dsbase, __bf16* bh, __bf16* bl, int tid) {
    int dblk  = (tid & 7) * 16;    // 16 consecutive f32 per lane
    int nrow0 = tid >> 3;          // 32 rows per pass
    #pragma unroll
    for (int p = 0; p < 4; ++p) {
        int n = nrow0 + p * 32;
        const float* ep = emb + (size_t)(kbase + n) * D_ + dsbase + dblk;
        alignas(16) __bf16 hh[16], ll[16];
        #pragma unroll
        for (int j = 0; j < 16; j += 4) {
            v4f v = *(const v4f*)(ep + j);
            #pragma unroll
            for (int c = 0; c < 4; ++c) {
                float f = v[c];
                __bf16 h = (__bf16)f;
                hh[j + c] = h;
                ll[j + c] = (__bf16)(f - (float)h);
            }
        }
        *(v8bf*)(bh + n * LDB + dblk)     = *(v8bf*)&hh[0];
        *(v8bf*)(bh + n * LDB + dblk + 8) = *(v8bf*)&hh[8];
        *(v8bf*)(bl + n * LDB + dblk)     = *(v8bf*)&ll[0];
        *(v8bf*)(bl + n * LDB + dblk + 8) = *(v8bf*)&ll[8];
    }
}

// ---------------- Kernel 2: fused GEMM + argmax + gather + loss partial ----------------
__global__ void __launch_bounds__(BLOCK)
vq_main(const float* __restrict__ x, const float* __restrict__ emb,
        const float* __restrict__ halfn, float* __restrict__ out_vals,
        int* __restrict__ out_idx, float* __restrict__ loss_part) {
    extern __shared__ __align__(16) char smem[];
    __bf16* a_hi  = (__bf16*)smem;                  // [BM][LDA]
    __bf16* a_lo  = a_hi + BM * LDA;                // [BM][LDA]
    __bf16* bbase = a_lo + BM * LDA;                // 2 ping-pong bufs of [BN][LDB] hi+lo
    float*  red_val = (float*)(bbase + 2 * BUFELE); // [BM][64]
    int*    red_idx = (int*)(red_val + BM * 64);    // [BM][64]
    __shared__ int   final_idx[BM];
    __shared__ float wsum[NWAVES];

    const int tid  = threadIdx.x;
    const int b    = blockIdx.x / (T_ / BM);
    const int t0   = (blockIdx.x % (T_ / BM)) * BM;
    const float* xblk = x + (size_t)b * D_ * T_ + t0;   // element (d,m) at d*T_+m

    // ---- Phase 1: transpose x tile into LDS as bf16 hi/lo, layout [m][d] ----
    {
        int m = tid & (BM - 1);
        #pragma unroll 4
        for (int d = tid >> 6; d < D_; d += 4) {
            float v = xblk[(size_t)d * T_ + m];
            __bf16 h = (__bf16)v;
            __bf16 l = (__bf16)(v - (float)h);
            a_hi[m * LDA + d] = h;
            a_lo[m * LDA + d] = l;
        }
    }

    const int wave   = tid >> 5;
    const int lane   = tid & 31;
    const int lhalf  = lane >> 4;     // 0/1 (lane half)
    const int lmod   = lane & 15;
    const int mbase  = (wave & 1) * 32;   // 2 m-groups of 32 rows
    const int ngroup = wave >> 1;         // 4 n-groups of 32 codes
    const int nbase  = ngroup * 32;

    float best_val[2][8];
    int   best_idx[2][8];
    #pragma unroll
    for (int mi = 0; mi < 2; ++mi)
        #pragma unroll
        for (int r = 0; r < 8; ++r) { best_val[mi][r] = -3.4e38f; best_idx[mi][r] = 0; }

    // Per-lane fragment offsets (contiguous k = 16*lhalf + s for both A and B)
    const __bf16* arow_hi0 = a_hi + (mbase +      lmod) * LDA + lhalf * 16;
    const __bf16* arow_hi1 = a_hi + (mbase + 16 + lmod) * LDA + lhalf * 16;
    const __bf16* arow_lo0 = a_lo + (mbase +      lmod) * LDA + lhalf * 16;
    const __bf16* arow_lo1 = a_lo + (mbase + 16 + lmod) * LDA + lhalf * 16;
    const int boff0 = (nbase +      lmod) * LDB + lhalf * 16;
    const int boff1 = (nbase + 16 + lmod) * LDB + lhalf * 16;

    v16bf fA_ah0, fA_ah1, fA_al0, fA_al1, fA_bh0, fA_bh1, fA_bl0, fA_bl1;
    v16bf fB_ah0, fB_ah1, fB_al0, fB_al1, fB_bh0, fB_bh1, fB_bl0, fB_bl1;

#define LOAD_FRAGSET(F, dg, dl, BH, BL)                         \
    F##_ah0 = ldA(arow_hi0 + (dg));                             \
    F##_ah1 = ldA(arow_hi1 + (dg));                             \
    F##_al0 = ldA(arow_lo0 + (dg));                             \
    F##_al1 = ldA(arow_lo1 + (dg));                             \
    F##_bh0 = *(const v16bf*)((BH) + boff0 + (dl));             \
    F##_bh1 = *(const v16bf*)((BH) + boff1 + (dl));             \
    F##_bl0 = *(const v16bf*)((BL) + boff0 + (dl));             \
    F##_bl1 = *(const v16bf*)((BL) + boff1 + (dl));

#define DO_WMMA12(F)                                            \
    acc00 = WMMA_BF16(F##_ah0, F##_bh0, acc00);                 \
    acc00 = WMMA_BF16(F##_ah0, F##_bl0, acc00);                 \
    acc00 = WMMA_BF16(F##_al0, F##_bh0, acc00);                 \
    acc01 = WMMA_BF16(F##_ah0, F##_bh1, acc01);                 \
    acc01 = WMMA_BF16(F##_ah0, F##_bl1, acc01);                 \
    acc01 = WMMA_BF16(F##_al0, F##_bh1, acc01);                 \
    acc10 = WMMA_BF16(F##_ah1, F##_bh0, acc10);                 \
    acc10 = WMMA_BF16(F##_ah1, F##_bl0, acc10);                 \
    acc10 = WMMA_BF16(F##_al1, F##_bh0, acc10);                 \
    acc11 = WMMA_BF16(F##_ah1, F##_bh1, acc11);                 \
    acc11 = WMMA_BF16(F##_ah1, F##_bl1, acc11);                 \
    acc11 = WMMA_BF16(F##_al1, F##_bh1, acc11);

    // ---- Phase 2: pipelined loop over codebook chunks (ping-pong B slices) ----
    stage_slice(emb, 0, 0, bbase, bbase + BN * LDB, tid);   // prologue: slice 0 -> buf 0
    __syncthreads();

    for (int kc = 0; kc < NKC; ++kc) {
        const int kbase = kc * BN;
        v8f acc00 = {0.f,0.f,0.f,0.f,0.f,0.f,0.f,0.f};
        v8f acc01 = acc00, acc10 = acc00, acc11 = acc00;

        #pragma unroll
        for (int si = 0; si < D_ / DS; ++si) {
            const int buf = si & 1;            // compile-time after unroll
            const int dsbase = si * DS;
            __bf16* curH = bbase + buf * BUFELE;
            __bf16* curL = curH + BN * LDB;

            // stage slice s+1 into the idle buffer; prefetch slice s+2 into cache
            if (!(kc == NKC - 1 && si == 3)) {
                int nk = (si == 3) ? kbase + BN : kbase;
                int nd = (si == 3) ? 0 : (si + 1) * DS;
                __bf16* nxtH = bbase + (buf ^ 1) * BUFELE;
                stage_slice(emb, nk, nd, nxtH, nxtH + BN * LDB, tid);
                int g = kc * 4 + si + 2;
                if (g < NKC * 4) {
                    int pk = (g >> 2) * BN, pd = (g & 3) * DS;
                    const float* pp = emb + (size_t)(pk + (tid >> 1)) * D_ + pd
                                    + (tid & 1) * 64;
                    __builtin_prefetch(pp, 0, 1);       // global_prefetch_b8
                    __builtin_prefetch(pp + 32, 0, 1);
                }
            }

            // 4 d-chunks, two-deep fragment pipeline (no WMMA->VALU WAR on frags)
            LOAD_FRAGSET(fA, dsbase +  0,  0, curH, curL)
            LOAD_FRAGSET(fB, dsbase + 32, 32, curH, curL)
            DO_WMMA12(fA)
            LOAD_FRAGSET(fA, dsbase + 64, 64, curH, curL)
            DO_WMMA12(fB)
            LOAD_FRAGSET(fB, dsbase + 96, 96, curH, curL)
            DO_WMMA12(fA)
            DO_WMMA12(fB)

            __syncthreads();   // one barrier per slice
        }

        // running argmax of (x.e - 0.5*||e||^2); lane sees codes kg0 and kg0+16
        const int   kg0  = kbase + nbase + lmod;
        const int   kg1  = kg0 + 16;
        const float adj0 = -halfn[kg0];
        const float adj1 = -halfn[kg1];
        #pragma unroll
        for (int r = 0; r < 8; ++r) {
            float s00 = acc00[r] + adj0, s01 = acc01[r] + adj1;
            float s10 = acc10[r] + adj0, s11 = acc11[r] + adj1;
            if (s00 > best_val[0][r]) { best_val[0][r] = s00; best_idx[0][r] = kg0; }
            if (s01 > best_val[0][r]) { best_val[0][r] = s01; best_idx[0][r] = kg1; }
            if (s10 > best_val[1][r]) { best_val[1][r] = s10; best_idx[1][r] = kg0; }
            if (s11 > best_val[1][r]) { best_val[1][r] = s11; best_idx[1][r] = kg1; }
        }
    }

    // ---- Phase 3: cross-lane / cross-wave argmax reduction ----
    #pragma unroll
    for (int mi = 0; mi < 2; ++mi)
        #pragma unroll
        for (int r = 0; r < 8; ++r) {
            int m    = mbase + mi * 16 + lhalf * 8 + r;  // C-layout: VGPR r -> M=r (+8 hi half)
            int slot = ngroup * 16 + lmod;               // 0..63
            red_val[m * 64 + slot] = best_val[mi][r];
            red_idx[m * 64 + slot] = best_idx[mi][r];
        }
    __syncthreads();
    if (tid < BM) {
        int m = tid;
        float bv = red_val[m * 64];
        int   bi = red_idx[m * 64];
        for (int s = 1; s < 64; ++s) {
            float v = red_val[m * 64 + s];
            if (v > bv) { bv = v; bi = red_idx[m * 64 + s]; }
        }
        final_idx[m] = bi;
        out_idx[b * T_ + t0 + m] = bi;
    }
    __syncthreads();

    // ---- Phase 4: gather winning rows to transposed output + loss partial ----
    {
        int m  = tid & (BM - 1);
        int bk = final_idx[m];
        const float* erow = emb + (size_t)bk * D_;
        float* orow = out_vals + (size_t)b * D_ * T_ + t0;
        float lsum = 0.f;
        #pragma unroll 4
        for (int d = tid >> 6; d < D_; d += 4) {
            float e  = erow[d];
            orow[(size_t)d * T_ + m] = e;
            float df = xblk[(size_t)d * T_ + m] - e;
            lsum += df * df;
        }
        #pragma unroll
        for (int off = 16; off > 0; off >>= 1) lsum += __shfl_xor(lsum, off, 32);
        if (lane == 0) wsum[wave] = lsum;
        __syncthreads();
        if (tid == 0) {
            float s = 0.f;
            #pragma unroll
            for (int i = 0; i < NWAVES; ++i) s += wsum[i];
            loss_part[blockIdx.x] = s;
        }
    }
}

// ---------------- Kernel 3: deterministic loss finalize ----------------
__global__ void vq_loss(const float* __restrict__ part, float* __restrict__ out_loss) {
    float s = 0.f;
    for (int i = 0; i < NBLOCKS; ++i) s += part[i];
    out_loss[0] = 2.0f * s / (float)((size_t)B_ * T_ * D_);
}

extern "C" void kernel_launch(void* const* d_in, const int* in_sizes, int n_in,
                              void* d_out, int out_size, void* d_ws, size_t ws_size,
                              hipStream_t stream) {
    (void)in_sizes; (void)n_in; (void)out_size; (void)ws_size;
    const float* x   = (const float*)d_in[0];   // [B, D, T] f32
    const float* emb = (const float*)d_in[1];   // [K, D]    f32

    float* out_vals = (float*)d_out;                                        // [B, D, T]
    int*   out_idx  = (int*)((float*)d_out + (size_t)B_ * D_ * T_);         // [B, T] int32
    float* out_loss = (float*)d_out + (size_t)B_ * D_ * T_ + (size_t)B_ * T_;

    float* halfn = (float*)d_ws;          // K_ floats
    float* parts = halfn + K_;            // NBLOCKS floats

    vq_norms<<<K_ / (BLOCK / 32), BLOCK, 0, stream>>>(emb, halfn);

    size_t smem = (size_t)(2 * BM * LDA + 4 * BN * LDB) * sizeof(__bf16)
                + (size_t)BM * 64 * (sizeof(float) + sizeof(int));   // ~298 KB of 320 KB
    hipFuncSetAttribute((const void*)vq_main,
                        hipFuncAttributeMaxDynamicSharedMemorySize, (int)smem);
    vq_main<<<NBLOCKS, BLOCK, smem, stream>>>(x, emb, halfn, out_vals, out_idx, parts);

    vq_loss<<<1, 1, 0, stream>>>(parts, out_loss);
}
